// DynamicRelationshipExtractor_17987323036464
// MI455X (gfx1250) — compile-verified, matrix operand order checked by reading, not run on hardware
//
#include <hip/hip_runtime.h>
#include <hip/hip_bf16.h>

typedef _Float16 half_t;
typedef __attribute__((ext_vector_type(16))) _Float16 v16h;
typedef __attribute__((ext_vector_type(8)))  _Float16 f16x8;
typedef __attribute__((ext_vector_type(4)))  _Float16 f16x4;
typedef __attribute__((ext_vector_type(8)))  float    v8f;
typedef __attribute__((ext_vector_type(4)))  float    f32x4;

#define D_DIM   128
#define H_DIM   64
#define P_NUM   2016
#define B_NUM   64
#define TILE_M  16
#define WAVES   4
#define CAT_K   1152

// f16 transposed-weight workspace layout (in half units)
#define W1T_OFF   0            // [8][64][128]   col-major per k  (65536)
#define W2T_OFF   65536        // [8][128][64]                    (65536)
#define PW2T_OFF  131072       // [128][64]                       (8192)
#define FW1T_OFF  139264       // [256][1152]                     (294912)
#define FW2T_OFF  434176       // [128][256]                      (32768)
#define WS_HALVES 466944

// per-wave dynamic LDS bytes:
// fA(8192) + fB(8192) + rnA(64) + rnB(64) + sFeats(4096) + sH(2048) + sCat(36864) = 59520
#define PER_WAVE_BYTES 59520

// ---------------- weight conversion / transposition (f32 -> f16 col-major) ----------------
__global__ void prep_weights(const float* __restrict__ opsW1, const float* __restrict__ opsW2,
                             const float* __restrict__ presW2, const float* __restrict__ fusW1,
                             const float* __restrict__ fusW2, half_t* __restrict__ ws) {
    int idx = blockIdx.x * blockDim.x + threadIdx.x;
    if (idx < 65536) {                       // w1T[k][n][d] = ops_W1[MLP_IDX[k]][d][n]
        int d = idx & 127, n = (idx >> 7) & 63, k = idx >> 13;
        int mi = (k <= 3) ? k : ((k <= 5) ? 4 : 5);
        ws[W1T_OFF + idx] = (half_t)opsW1[(mi * 128 + d) * 64 + n];
    } else if (idx < 131072) {               // w2T[k][n][h] = ops_W2[MLP_IDX[k]][h][n]
        int i2 = idx - 65536;
        int h = i2 & 63, n = (i2 >> 6) & 127, k = i2 >> 13;
        int mi = (k <= 3) ? k : ((k <= 5) ? 4 : 5);
        ws[W2T_OFF + i2] = (half_t)opsW2[(mi * 64 + h) * 128 + n];
    } else if (idx < 139264) {               // presW2T[n][h]
        int i2 = idx - 131072;
        int h = i2 & 63, n = i2 >> 6;
        ws[PW2T_OFF + i2] = (half_t)presW2[h * 128 + n];
    } else if (idx < 434176) {               // fusW1T[n][r]  (1152 rows)
        int i2 = idx - 139264;
        int r = i2 % 1152, n = i2 / 1152;
        ws[FW1T_OFF + i2] = (half_t)fusW1[r * 256 + n];
    } else if (idx < WS_HALVES) {            // fusW2T[n][r]  (256 rows)
        int i2 = idx - 434176;
        int r = i2 & 255, n = i2 >> 8;
        ws[FW2T_OFF + i2] = (half_t)fusW2[r * 128 + n];
    }
}

// ---------------- wave-local LDS sync ----------------
// All LDS buffers are private to one wave; producers/consumers are lanes of the
// same wave. LDS ops complete in-order per wave (DScnt), so we only need a
// compiler scheduling fence around an explicit DScnt drain — no s_barrier.
__device__ __forceinline__ void wave_sync() {
    __builtin_amdgcn_wave_barrier();
    asm volatile("s_wait_dscnt 0x0" ::: "memory");
    __builtin_amdgcn_wave_barrier();
}

// ---------------- WMMA helpers ----------------
// 16-bit A/B fragment (16x32 / 32x16): lane's 16 halves are two contiguous
// 16-byte runs at K = k0 + g*8 and K = k0 + 16 + g*8 (g = lane>>4).
__device__ __forceinline__ v16h load_frag(const half_t* p) {
    f16x8 lo = *(const f16x8*)p;
    f16x8 hi = *(const f16x8*)(p + 16);
    v16h r;
#pragma unroll
    for (int i = 0; i < 8; ++i) { r[i] = lo[i]; r[i + 8] = hi[i]; }
    return r;
}

__device__ __forceinline__ v8f wmma16(v16h a, v16h b, v8f c) {
    return __builtin_amdgcn_wmma_f32_16x16x32_f16(false, a, false, b, (short)0, c, false, false);
}

// ---------------- fused main kernel ----------------
__global__ __launch_bounds__(128, 1) void dre_fused_kernel(
    const float* __restrict__ x, const int* __restrict__ presence,
    const int* __restrict__ idx_i, const int* __restrict__ idx_j,
    const float* __restrict__ ops_b1, const float* __restrict__ ops_b2,
    const float* __restrict__ pres_W1, const float* __restrict__ pres_b1,
    const float* __restrict__ pres_b2,
    const float* __restrict__ fus_b1, const float* __restrict__ fus_b2,
    const half_t* __restrict__ w1T, const half_t* __restrict__ w2T,
    const half_t* __restrict__ pW2T, const half_t* __restrict__ fW1T,
    const half_t* __restrict__ fW2T,
    float* __restrict__ out)
{
    extern __shared__ char smem[];
    const int wave  = threadIdx.x >> 5;
    const int lane  = threadIdx.x & 31;
    const int mlane = lane & 15;
    const int g     = lane >> 4;

    char*   wbase  = smem + wave * PER_WAVE_BYTES;
    float*  fA     = (float*)wbase;            // 16x128 f32
    float*  fB     = fA + 16 * D_DIM;          // 16x128 f32
    float*  rnA    = fB + 16 * D_DIM;          // 16
    float*  rnB    = rnA + 16;                 // 16
    half_t* sFeats = (half_t*)(rnB + 16);      // 16x128 f16
    half_t* sH     = sFeats + 16 * D_DIM;      // 16x64  f16
    half_t* sCat   = sH + 16 * H_DIM;          // 16x1152 f16 (full concat row block)
    half_t* sHid   = (half_t*)fA;              // 16x256 f16 (reuses fA/fB space after k loop)

    const int tile    = blockIdx.x * WAVES + wave;       // 8064 tiles total
    const int rowBase = tile * TILE_M;
    const int b       = rowBase / P_NUM;                  // tiles never cross b (2016/16=126)
    const int p0      = rowBase - b * P_NUM;

    __builtin_prefetch(fW1T, 0, 1);
    __builtin_prefetch(fW2T, 0, 1);

    // ---- Phase 1: gather A/Bv rows, compute inverse norms ----
    for (int m = 0; m < TILE_M; ++m) {
        int p  = p0 + m;
        int ii = idx_i[p], jj = idx_j[p];
        const float* ra = x + ((size_t)(b * B_NUM + ii)) * D_DIM + lane * 4;
        const float* rb = x + ((size_t)(b * B_NUM + jj)) * D_DIM + lane * 4;
        f32x4 av = *(const f32x4*)ra;
        f32x4 bv = *(const f32x4*)rb;
        *(f32x4*)&fA[m * D_DIM + lane * 4] = av;
        *(f32x4*)&fB[m * D_DIM + lane * 4] = bv;
        float sa = av[0]*av[0] + av[1]*av[1] + av[2]*av[2] + av[3]*av[3];
        float sb = bv[0]*bv[0] + bv[1]*bv[1] + bv[2]*bv[2] + bv[3]*bv[3];
#pragma unroll
        for (int off = 16; off; off >>= 1) {
            sa += __shfl_xor(sa, off, 32);
            sb += __shfl_xor(sb, off, 32);
        }
        if (lane == 0) {
            rnA[m] = 1.f / (sqrtf(sa) + 1e-6f);
            rnB[m] = 1.f / (sqrtf(sb) + 1e-6f);
        }
    }

    // ---- Phase 2: presence hidden h_p = relu(flags @ pres_W1 + pres_b1) ----
    for (int m = 0; m < TILE_M; ++m) {
        int p  = p0 + m;
        int ii = idx_i[p], jj = idx_j[p];
        float pa = (float)presence[b * B_NUM + ii];
        float pb = (float)presence[b * B_NUM + jj];
        float f0 = pa * pb, f1 = pa * (1.f - pb), f2 = (1.f - pa) * pb, f3 = (1.f - pa) * (1.f - pb);
#pragma unroll
        for (int u = 0; u < 2; ++u) {
            int jc = lane * 2 + u;
            float h = f0 * pres_W1[jc] + f1 * pres_W1[64 + jc] +
                      f2 * pres_W1[128 + jc] + f3 * pres_W1[192 + jc] + pres_b1[jc];
            sH[m * H_DIM + jc] = (half_t)fmaxf(h, 0.f);
        }
    }
    wave_sync();

    // ---- Phase 2b: ptok = h_p(16x64) @ presW2 + pres_b2 -> sCat[:, 1024:1152] ----
    {
        v8f acc[8];
#pragma unroll
        for (int t = 0; t < 8; ++t) {
            float bv = pres_b2[t * 16 + mlane];
#pragma unroll
            for (int r = 0; r < 8; ++r) acc[t][r] = bv;
        }
#pragma unroll
        for (int kb = 0; kb < 2; ++kb) {
            v16h af = load_frag(sH + mlane * H_DIM + kb * 32 + g * 8);
#pragma unroll
            for (int t = 0; t < 8; ++t) {
                int n = t * 16 + mlane;
                v16h bf = load_frag(pW2T + (size_t)n * H_DIM + kb * 32 + g * 8);
                acc[t] = wmma16(af, bf, acc[t]);
            }
        }
#pragma unroll
        for (int t = 0; t < 8; ++t)
#pragma unroll
            for (int r = 0; r < 8; ++r)
                sCat[(r + g * 8) * CAT_K + 1024 + t * 16 + mlane] = (half_t)acc[t][r];
    }
    wave_sync();

    // ---- Phase 3: 8 feature maps, each: feats -> MLP -> tok_k into sCat[:, k*128:...] ----
    for (int k = 0; k < 8; ++k) {
        // feats_k -> sFeats (f16)
        for (int m = 0; m < TILE_M; ++m) {
            int c = lane * 4;
            f32x4 av = *(f32x4*)&fA[m * D_DIM + c];
            f32x4 bv = *(f32x4*)&fB[m * D_DIM + c];
            float ia = rnA[m], ib = rnB[m];
            f16x4 hv;
#pragma unroll
            for (int u = 0; u < 4; ++u) {
                float a = av[u], bb = bv[u], f;
                switch (k) {
                    case 0:  f = a * bb; break;
                    case 1:  f = a + bb; break;
                    case 2:  f = (a * ia) * (bb * ib); break;
                    case 3:  f = fabsf(a - bb); break;
                    case 4:  f = a - bb; break;
                    case 5:  f = bb - a; break;
                    case 6:  f = a / (bb + 1e-6f); break;
                    default: f = bb / (a + 1e-6f); break;
                }
                hv[u] = (half_t)f;
            }
            *(f16x4*)&sFeats[m * D_DIM + c] = hv;
        }
        wave_sync();

        int mi = (k <= 3) ? k : ((k <= 5) ? 4 : 5);

        // layer 1: h = relu(feats(16x128) @ W1_k(128x64) + b1)
        v8f accH[4];
#pragma unroll
        for (int t = 0; t < 4; ++t) {
            float bv = ops_b1[mi * 64 + t * 16 + mlane];
#pragma unroll
            for (int r = 0; r < 8; ++r) accH[t][r] = bv;
        }
#pragma unroll
        for (int kb = 0; kb < 4; ++kb) {
            v16h af = load_frag(sFeats + mlane * D_DIM + kb * 32 + g * 8);
#pragma unroll
            for (int t = 0; t < 4; ++t) {
                int n = t * 16 + mlane;
                v16h bf = load_frag(w1T + ((size_t)(k * 64 + n)) * D_DIM + kb * 32 + g * 8);
                accH[t] = wmma16(af, bf, accH[t]);
            }
        }
#pragma unroll
        for (int t = 0; t < 4; ++t)
#pragma unroll
            for (int r = 0; r < 8; ++r)
                sH[(r + g * 8) * H_DIM + t * 16 + mlane] = (half_t)fmaxf(accH[t][r], 0.f);
        wave_sync();

        // layer 2: tok_k = h(16x64) @ W2_k(64x128) + b2  -> sCat columns k*128..
        v8f accT[8];
#pragma unroll
        for (int t = 0; t < 8; ++t) {
            float bv = ops_b2[mi * 128 + t * 16 + mlane];
#pragma unroll
            for (int r = 0; r < 8; ++r) accT[t][r] = bv;
        }
#pragma unroll
        for (int kb = 0; kb < 2; ++kb) {
            v16h af = load_frag(sH + mlane * H_DIM + kb * 32 + g * 8);
#pragma unroll
            for (int t = 0; t < 8; ++t) {
                int n = t * 16 + mlane;
                v16h bf = load_frag(w2T + ((size_t)(k * 128 + n)) * H_DIM + kb * 32 + g * 8);
                accT[t] = wmma16(af, bf, accT[t]);
            }
        }
#pragma unroll
        for (int t = 0; t < 8; ++t)
#pragma unroll
            for (int r = 0; r < 8; ++r)
                sCat[(r + g * 8) * CAT_K + k * 128 + t * 16 + mlane] = (half_t)accT[t][r];
        wave_sync();
    }

    // ---- Phase 4: fusion layer 1 over full K=1152 (tok blocks + ptok block) ----
    // hid = relu(cat(16x1152) @ fus_W1 + fus_b1) -> sHid (16x256 f16, reuses fA/fB)
    for (int hh = 0; hh < 2; ++hh) {
        v8f accF[8];
#pragma unroll
        for (int t = 0; t < 8; ++t) {
            float bv = fus_b1[(hh * 8 + t) * 16 + mlane];
#pragma unroll
            for (int r = 0; r < 8; ++r) accF[t][r] = bv;
        }
        __builtin_prefetch(fW1T + (size_t)((hh * 8) * 16 + mlane) * CAT_K, 0, 1);
#pragma unroll 4
        for (int kb = 0; kb < 36; ++kb) {
            v16h af = load_frag(sCat + mlane * CAT_K + kb * 32 + g * 8);
#pragma unroll
            for (int t = 0; t < 8; ++t) {
                int n = (hh * 8 + t) * 16 + mlane;
                v16h bf = load_frag(fW1T + (size_t)n * CAT_K + kb * 32 + g * 8);
                accF[t] = wmma16(af, bf, accF[t]);
            }
        }
#pragma unroll
        for (int t = 0; t < 8; ++t)
#pragma unroll
            for (int r = 0; r < 8; ++r)
                sHid[(r + g * 8) * 256 + hh * 128 + t * 16 + mlane] = (half_t)fmaxf(accF[t][r], 0.f);
    }
    wave_sync();

    // ---- Phase 5: out = hid(16x256) @ fus_W2 + fus_b2 ----
    v8f accO[8];
#pragma unroll
    for (int t = 0; t < 8; ++t) {
        float bv = fus_b2[t * 16 + mlane];
#pragma unroll
        for (int r = 0; r < 8; ++r) accO[t][r] = bv;
    }
#pragma unroll
    for (int kb = 0; kb < 8; ++kb) {
        v16h af = load_frag(sHid + mlane * 256 + kb * 32 + g * 8);
#pragma unroll
        for (int t = 0; t < 8; ++t) {
            int n = t * 16 + mlane;
            v16h bf = load_frag(fW2T + (size_t)n * 256 + kb * 32 + g * 8);
            accO[t] = wmma16(af, bf, accO[t]);
        }
    }

#pragma unroll
    for (int t = 0; t < 8; ++t)
#pragma unroll
        for (int r = 0; r < 8; ++r) {
            int m = r + g * 8;
            out[(size_t)(rowBase + m) * D_DIM + t * 16 + mlane] = accO[t][r];
        }
}

// ---------------- host launch ----------------
extern "C" void kernel_launch(void* const* d_in, const int* in_sizes, int n_in,
                              void* d_out, int out_size, void* d_ws, size_t ws_size,
                              hipStream_t stream) {
    const float* x        = (const float*)d_in[0];
    const int*   presence = (const int*)d_in[1];
    const int*   idx_i    = (const int*)d_in[2];
    const int*   idx_j    = (const int*)d_in[3];
    const float* ops_W1   = (const float*)d_in[4];
    const float* ops_b1   = (const float*)d_in[5];
    const float* ops_W2   = (const float*)d_in[6];
    const float* ops_b2   = (const float*)d_in[7];
    const float* pres_W1  = (const float*)d_in[8];
    const float* pres_b1  = (const float*)d_in[9];
    const float* pres_W2  = (const float*)d_in[10];
    const float* pres_b2  = (const float*)d_in[11];
    const float* fus_W1   = (const float*)d_in[12];
    const float* fus_b1   = (const float*)d_in[13];
    const float* fus_W2   = (const float*)d_in[14];
    const float* fus_b2   = (const float*)d_in[15];
    half_t* ws = (half_t*)d_ws;

    // convert/transpose all weights to f16 column-major in workspace (~0.9 MB)
    prep_weights<<<(WS_HALVES + 255) / 256, 256, 0, stream>>>(
        ops_W1, ops_W2, pres_W2, fus_W1, fus_W2, ws);

    // fused forward: 8064 tiles of 16 rows, 4 waves/block (238 KB LDS/block)
    size_t smem = (size_t)PER_WAVE_BYTES * WAVES;
    dre_fused_kernel<<<2016, 128, smem, stream>>>(
        x, presence, idx_i, idx_j,
        ops_b1, ops_b2, pres_W1, pres_b1, pres_b2, fus_b1, fus_b2,
        ws + W1T_OFF, ws + W2T_OFF, ws + PW2T_OFF, ws + FW1T_OFF, ws + FW2T_OFF,
        (float*)d_out);
}
